// ConditionedMLP_11192684773731
// MI455X (gfx1250) — compile-verified, matrix operand order checked by reading, not run on hardware
//
#include <hip/hip_runtime.h>

// ---------------------------------------------------------------------------
// ConditionedMLP on gfx1250 (MI455X), restructured as bf16 WMMA GEMMs.
//
//   per layer:  h  = x @ W.T + b                       (M=2048,N=256,K=128/256)
//               x' = relu( Z @ U.T + Fb[65536:] )      (M=2048,N=256,K=16704)
//   where Z[b] = concat(h[b] (x) side[b], h[b], side[b])  -- built on the fly
//   and   U[n] = concat(Fw[n*256:(n+1)*256,:].flat, Fb[n*256:+256], Fw[65536+n,:])
//   output:     out = h3 @ Wout.T + bout               (M=2048,N=64, K=256)
// ---------------------------------------------------------------------------

typedef __attribute__((ext_vector_type(16))) __bf16 v16bf;
typedef __attribute__((ext_vector_type(8)))  float  v8f;

#define KCOND 16704   // 256*64 + 256 + 64

__device__ __forceinline__ v8f wmma_bf16(v16bf a, v16bf b, v8f c) {
  // D = A(16x32) * B(32x16) + C, f32 accumulate
  return __builtin_amdgcn_wmma_f32_16x16x32_bf16(false, a, false, b,
                                                 (short)0, c, false, false);
}

// f32 -> bf16 elementwise convert
__global__ __launch_bounds__(256) void cvt_f32_bf16(const float* __restrict__ in,
                                                    __bf16* __restrict__ out, int n) {
  int stride = gridDim.x * blockDim.x;
  for (int i = blockIdx.x * blockDim.x + threadIdx.x; i < n; i += stride)
    out[i] = (__bf16)in[i];
}

// Gather-convert the conditioned-layer weight matrix U (256 x 16704, row major)
__global__ __launch_bounds__(256) void pack_u(const float* __restrict__ Fw,
                                              const float* __restrict__ Fb,
                                              __bf16* __restrict__ U) {
  int n = blockIdx.x;  // 0..255 output feature
  for (int m = threadIdx.x; m < KCOND; m += 256) {
    float v;
    if (m < 16384)      v = Fw[n * 16384 + m];                 // Fw[n*256+j, k], m=j*64+k
    else if (m < 16640) v = Fb[n * 256 + (m - 16384)];         // Fb[n*256+j]
    else                v = Fw[4194304 + n * 64 + (m - 16640)];// Fw[65536+n, k]
    U[(size_t)n * KCOND + m] = (__bf16)v;
  }
}

__device__ __forceinline__ void store_out(v8f c, int row0, int col, int N,
                                          const float* __restrict__ bias, int relu,
                                          __bf16* __restrict__ Obf,
                                          float* __restrict__ Of) {
  float bv = bias[col];
#pragma unroll
  for (int r = 0; r < 8; ++r) {
    float v = c[r] + bv;
    if (relu) v = fmaxf(v, 0.0f);
    if (Obf) Obf[(size_t)(row0 + r) * N + col] = (__bf16)v;
    else     Of [(size_t)(row0 + r) * N + col] = v;
  }
}

// Generic C = A(MxK,bf16) * Bt(NxK,bf16).T + bias, optional relu, bf16 or f32 out.
// Block = 256 thr = 8 waves (4 M x 2 N), wave tile 32x32, block tile 128x64.
__global__ __launch_bounds__(256) void gemm_bf16(
    const __bf16* __restrict__ A, const __bf16* __restrict__ Bt,
    const float* __restrict__ bias, __bf16* __restrict__ Obf,
    float* __restrict__ Of, int M, int N, int K, int relu) {
  const int lane = threadIdx.x & 31, wave = threadIdx.x >> 5;
  const int wy = wave & 3, wx = wave >> 2;
  const int lr = lane & 15, lhalf = lane >> 4;
  const int m0 = blockIdx.y * 128 + wy * 32;
  const int n0 = blockIdx.x * 64 + wx * 32;

  const __bf16* a0p = A + (size_t)(m0 + lr) * K + lhalf * 16;
  const __bf16* a1p = a0p + (size_t)16 * K;
  const __bf16* b0p = Bt + (size_t)(n0 + lr) * K + lhalf * 16;
  const __bf16* b1p = b0p + (size_t)16 * K;

  v8f c00 = {}, c01 = {}, c10 = {}, c11 = {};
  for (int k = 0; k < K; k += 32) {
    v16bf a0 = *(const v16bf*)(a0p + k);
    v16bf a1 = *(const v16bf*)(a1p + k);
    v16bf b0 = *(const v16bf*)(b0p + k);
    v16bf b1 = *(const v16bf*)(b1p + k);
    c00 = wmma_bf16(a0, b0, c00);
    c01 = wmma_bf16(a0, b1, c01);
    c10 = wmma_bf16(a1, b0, c10);
    c11 = wmma_bf16(a1, b1, c11);
  }
  const int grow = m0 + 8 * lhalf;
  store_out(c00, grow,      n0 + lr,      N, bias, relu, Obf, Of);
  store_out(c01, grow,      n0 + 16 + lr, N, bias, relu, Obf, Of);
  store_out(c10, grow + 16, n0 + lr,      N, bias, relu, Obf, Of);
  store_out(c11, grow + 16, n0 + 16 + lr, N, bias, relu, Obf, Of);
}

// Conditioned layer GEMM: Out = relu(Z @ U.T + bias), Z built on the fly from
// LDS-staged h (64x256) and side (64x64) tiles. M=2048, N=256, K=16704.
// Block = 256 thr = 8 waves (2 M x 4 N), wave tile 32x32, block tile 64x128.
__global__ __launch_bounds__(256) void cond_gemm(
    const __bf16* __restrict__ H, const __bf16* __restrict__ S,
    const __bf16* __restrict__ U, const float* __restrict__ bias,
    __bf16* __restrict__ Out) {
  __shared__ __align__(32) __bf16 hs[64 * 256];  // 32 KB
  __shared__ __align__(32) __bf16 ss[64 * 64];   //  8 KB
  const int tid = threadIdx.x;
  const int bm0 = blockIdx.y * 64;
  const int bn0 = blockIdx.x * 128;

  {  // cooperative stage: 16B chunks
    const uint4* gh = (const uint4*)(H + (size_t)bm0 * 256);
    uint4* lh = (uint4*)hs;
#pragma unroll
    for (int i = 0; i < 8; ++i) lh[tid + 256 * i] = gh[tid + 256 * i];   // 2048 chunks
    const uint4* gs = (const uint4*)(S + (size_t)bm0 * 64);
    uint4* ls = (uint4*)ss;
#pragma unroll
    for (int i = 0; i < 2; ++i) ls[tid + 256 * i] = gs[tid + 256 * i];   // 512 chunks
  }
  __syncthreads();

  const int lane = tid & 31, wave = tid >> 5;
  const int wy = wave & 1, wx = wave >> 1;
  const int lr = lane & 15, lhalf = lane >> 4;
  const int m0 = wy * 32;
  const int n0 = bn0 + wx * 32;

  const __bf16* h0 = hs + (m0 + lr) * 256;
  const __bf16* h1 = h0 + 16 * 256;
  const __bf16* s0 = ss + (m0 + lr) * 64;
  const __bf16* s1 = s0 + 16 * 64;
  const __bf16* b0p = U + (size_t)(n0 + lr) * KCOND + lhalf * 16;
  const __bf16* b1p = b0p + (size_t)16 * KCOND;

  v8f c00 = {}, c01 = {}, c10 = {}, c11 = {};

  // Segment 1: K in [0,16384) -> Z[k0] = h[k0>>6] * side[k0&63]
  for (int k = 0; k < 16384; k += 32) {
    const int j  = k >> 6;                        // uniform across both lane halves
    const int kk = (k & 63) + lhalf * 16;
    v16bf hb0 = (v16bf)h0[j];
    v16bf hb1 = (v16bf)h1[j];
    v16bf a0 = *(const v16bf*)(s0 + kk) * hb0;    // v_pk_mul_bf16, co-exec with WMMA
    v16bf a1 = *(const v16bf*)(s1 + kk) * hb1;
    v16bf b0 = *(const v16bf*)(b0p + k);
    v16bf b1 = *(const v16bf*)(b1p + k);
    __builtin_prefetch(b0p + k + 1024, 0, 1);
    __builtin_prefetch(b1p + k + 1024, 0, 1);
    c00 = wmma_bf16(a0, b0, c00);
    c01 = wmma_bf16(a0, b1, c01);
    c10 = wmma_bf16(a1, b0, c10);
    c11 = wmma_bf16(a1, b1, c11);
  }
  // Segment 2: K in [16384,16640) -> Z[k0] = h[k0-16384]
  for (int k = 16384; k < 16640; k += 32) {
    const int kk = (k - 16384) + lhalf * 16;
    v16bf a0 = *(const v16bf*)(h0 + kk);
    v16bf a1 = *(const v16bf*)(h1 + kk);
    v16bf b0 = *(const v16bf*)(b0p + k);
    v16bf b1 = *(const v16bf*)(b1p + k);
    c00 = wmma_bf16(a0, b0, c00);
    c01 = wmma_bf16(a0, b1, c01);
    c10 = wmma_bf16(a1, b0, c10);
    c11 = wmma_bf16(a1, b1, c11);
  }
  // Segment 3: K in [16640,16704) -> Z[k0] = side[k0-16640]
  for (int k = 16640; k < 16704; k += 32) {
    const int kk = (k - 16640) + lhalf * 16;
    v16bf a0 = *(const v16bf*)(s0 + kk);
    v16bf a1 = *(const v16bf*)(s1 + kk);
    v16bf b0 = *(const v16bf*)(b0p + k);
    v16bf b1 = *(const v16bf*)(b1p + k);
    c00 = wmma_bf16(a0, b0, c00);
    c01 = wmma_bf16(a0, b1, c01);
    c10 = wmma_bf16(a1, b0, c10);
    c11 = wmma_bf16(a1, b1, c11);
  }

  const int grow = bm0 + m0 + 8 * lhalf;
  store_out(c00, grow,      n0 + lr,      256, bias, 1, Out, nullptr);
  store_out(c01, grow,      n0 + 16 + lr, 256, bias, 1, Out, nullptr);
  store_out(c10, grow + 16, n0 + lr,      256, bias, 1, Out, nullptr);
  store_out(c11, grow + 16, n0 + 16 + lr, 256, bias, 1, Out, nullptr);
}

extern "C" void kernel_launch(void* const* d_in, const int* in_sizes, int n_in,
                              void* d_out, int out_size, void* d_ws, size_t ws_size,
                              hipStream_t stream) {
  (void)in_sizes; (void)n_in; (void)out_size; (void)ws_size;
  // setup_inputs order: side, x, (W,b,Fw,Fb) x3, Wout, bout
  const float* side = (const float*)d_in[0];
  const float* x    = (const float*)d_in[1];
  const float* Wl[3] = {(const float*)d_in[2], (const float*)d_in[6],  (const float*)d_in[10]};
  const float* bl[3] = {(const float*)d_in[3], (const float*)d_in[7],  (const float*)d_in[11]};
  const float* Fw[3] = {(const float*)d_in[4], (const float*)d_in[8],  (const float*)d_in[12]};
  const float* Fb[3] = {(const float*)d_in[5], (const float*)d_in[9],  (const float*)d_in[13]};
  const float* Wout = (const float*)d_in[14];
  const float* bout = (const float*)d_in[15];

  char* ws = (char*)d_ws;
  size_t off = 0;
  auto salloc = [&](size_t bytes) -> void* {
    void* p = ws + off;
    off += (bytes + 255) & ~(size_t)255;
    return p;
  };
  __bf16* xb    = (__bf16*)salloc(2048ull * 128 * 2);
  __bf16* sideb = (__bf16*)salloc(2048ull * 64 * 2);
  __bf16* Wb[3];
  Wb[0] = (__bf16*)salloc(256ull * 128 * 2);
  Wb[1] = (__bf16*)salloc(256ull * 256 * 2);
  Wb[2] = (__bf16*)salloc(256ull * 256 * 2);
  __bf16* Woutb = (__bf16*)salloc(64ull * 256 * 2);
  __bf16* U[3];
  for (int i = 0; i < 3; ++i) U[i] = (__bf16*)salloc((size_t)256 * KCOND * 2);
  __bf16* hbuf = (__bf16*)salloc(2048ull * 256 * 2);
  __bf16* actA = (__bf16*)salloc(2048ull * 256 * 2);
  __bf16* actB = (__bf16*)salloc(2048ull * 256 * 2);
  // total ~30 MB

  // bf16 conversions + weight packing
  cvt_f32_bf16<<<128, 256, 0, stream>>>(x,    xb,    2048 * 128);
  cvt_f32_bf16<<<64,  256, 0, stream>>>(side, sideb, 2048 * 64);
  cvt_f32_bf16<<<32,  256, 0, stream>>>(Wl[0], Wb[0], 256 * 128);
  cvt_f32_bf16<<<64,  256, 0, stream>>>(Wl[1], Wb[1], 256 * 256);
  cvt_f32_bf16<<<64,  256, 0, stream>>>(Wl[2], Wb[2], 256 * 256);
  cvt_f32_bf16<<<16,  256, 0, stream>>>(Wout, Woutb, 64 * 256);
  for (int i = 0; i < 3; ++i)
    pack_u<<<256, 256, 0, stream>>>(Fw[i], Fb[i], U[i]);

  // 3 conditioned layers
  const __bf16* cur = xb;
  int kin = 128;
  __bf16* acts[3] = {actA, actB, actA};
  for (int i = 0; i < 3; ++i) {
    gemm_bf16<<<dim3(256 / 64, 2048 / 128), 256, 0, stream>>>(
        cur, Wb[i], bl[i], hbuf, nullptr, 2048, 256, kin, 0);
    cond_gemm<<<dim3(256 / 128, 2048 / 64), 256, 0, stream>>>(
        hbuf, sideb, U[i], Fb[i] + 65536, acts[i]);
    cur = acts[i];
    kin = 256;
  }
  // output projection -> f32 d_out (2048 x 64)
  gemm_bf16<<<dim3(1, 2048 / 128), 256, 0, stream>>>(
      cur, Woutb, bout, nullptr, (float*)d_out, 2048, 64, 256, 0);
}
